// MoERanPAC_5480378269789
// MI455X (gfx1250) — matrix operand order repaired and checked by reading, not compiled, not run on hardware
//
#include <hip/hip_runtime.h>
#include <cstdint>
#include <cstddef>

// ---------- types ----------
typedef __attribute__((ext_vector_type(16))) __bf16 v16bf;
typedef __attribute__((ext_vector_type(8)))  __bf16 v8bf;
typedef __attribute__((ext_vector_type(8)))  float  v8f;
typedef __attribute__((ext_vector_type(4)))  unsigned int v4u;
typedef __attribute__((ext_vector_type(8)))  int v8i;
typedef __attribute__((ext_vector_type(4)))  int v4i;

union Frag16 { v16bf v; v8bf h[2]; unsigned short u[16]; };

#if __has_builtin(__builtin_amdgcn_tensor_load_to_lds) && __has_builtin(__builtin_amdgcn_s_wait_tensorcnt)
#define USE_TDM 1
#else
#define USE_TDM 0
#endif

static __device__ __forceinline__ unsigned short f2b_rne(float f) {
    unsigned int u = __float_as_uint(f);
    unsigned int r = u + 0x7FFFu + ((u >> 16) & 1u);
    return (unsigned short)(r >> 16);
}
static __device__ __forceinline__ float b2f(unsigned short h) {
    return __uint_as_float(((unsigned int)h) << 16);
}
static __device__ __forceinline__ float out_ld(const float v) { return v; }
static __device__ __forceinline__ float out_ld(const unsigned short v) { return b2f(v); }
static __device__ __forceinline__ void out_st(float& d, float v) { d = v; }
static __device__ __forceinline__ void out_st(unsigned short& d, float v) { d = f2b_rne(v); }

// ---------- TDM: async 2D tile load global -> LDS (D# per CDNA5 ISA 8.3/8.4) ----------
// data_size = 2 bytes; dims/strides in elements. OOB rows zero-fill LDS.
static __device__ __forceinline__ void tdm_load_tile_2d(
    unsigned lds_off, const unsigned short* g,
    unsigned tile_w, unsigned tile_h,
    unsigned tens_w, unsigned tens_h,
    unsigned long long stride_elems)
{
#if USE_TDM
    unsigned long long ga = (unsigned long long)(uintptr_t)g;
    v4u g0;
    g0.x = 1u;                                              // count=1, user mode, no gather
    g0.y = lds_off;                                         // LDS byte address
    g0.z = (unsigned)(ga & 0xFFFFFFFFu);                    // global_addr[31:0]
    g0.w = (unsigned)((ga >> 32) & 0x01FFFFFFu) | (2u << 30); // global_addr[56:32], type=2
    v8i g1;
    g1[0] = (int)(1u << 16);                                // data_size=1 (2B), wg_mask=0
    g1[1] = (int)((tens_w & 0xFFFFu) << 16);                // tensor_dim0[15:0] @ bits 63:48
    g1[2] = (int)(((tens_w >> 16) & 0xFFFFu) | ((tens_h & 0xFFFFu) << 16));
    g1[3] = (int)(((tens_h >> 16) & 0xFFFFu) | ((tile_w & 0xFFFFu) << 16));
    g1[4] = (int)(tile_h & 0xFFFFu);                        // tile_dim1; tile_dim2=0
    g1[5] = (int)(unsigned)(stride_elems & 0xFFFFFFFFu);    // tensor_dim0_stride[31:0]
    g1[6] = (int)(unsigned)((stride_elems >> 32) & 0xFFFFu);// stride[47:32]; dim1_stride=0
    g1[7] = 0;
    v4i z4 = {0, 0, 0, 0};
#if defined(__clang_major__) && (__clang_major__ >= 23)
    v8i z8 = {0, 0, 0, 0, 0, 0, 0, 0};
    __builtin_amdgcn_tensor_load_to_lds(g0, g1, z4, z4, z8, 0);
#else
    __builtin_amdgcn_tensor_load_to_lds(g0, g1, z4, z4, 0);
#endif
#else
    (void)lds_off; (void)g; (void)tile_w; (void)tile_h;
    (void)tens_w; (void)tens_h; (void)stride_elems;
#endif
}

// ---------- fp32 -> bf16 conversion ----------
__global__ void cvt_f32_bf16(const float* __restrict__ in,
                             unsigned short* __restrict__ out, long n) {
    long i = (long)blockIdx.x * blockDim.x + threadIdx.x;
    long stride = (long)gridDim.x * blockDim.x;
    for (; i < n; i += stride) out[i] = f2b_rne(in[i]);
}

// ---------- generic bf16 WMMA GEMM:  C = alpha * A(MxK) * B(NxK)^T [+bias][+C] ----------
// TDM double-buffered LDS staging: A tile 64x32, B tile 256x32 per k-step.
// Block: 256 threads = 8 waves (2x4), wave tile 32(M) x 64(N) = 2x4 WMMA frags,
// so 8 WMMAs per k-step reuse 2 A-frags -> high WMMA/LDS-byte ratio.
template <typename OutT>
__global__ __launch_bounds__(256)
void gemm_bf16_wmma(const unsigned short* __restrict__ A, int lda,
                    const unsigned short* __restrict__ B, int ldb,
                    OutT* __restrict__ C, long ldc,
                    const float* __restrict__ bias,
                    int M, int N, int K, float alpha, int accumulate) {
    __shared__ __align__(128) unsigned short As[2][64][32];    //  8 KB
    __shared__ __align__(128) unsigned short Bs[2][256][32];   // 32 KB

    const int lane = threadIdx.x & 31;
    const int wave = threadIdx.x >> 5;
    const int wm = wave >> 2;          // 0..1
    const int wn = wave & 3;           // 0..3
    const int lm = lane & 15;
    const int hi = lane >> 4;          // 0/1 half-wave
    const int hl = hi * 8;

    const int bm0 = blockIdx.y * 64;
    const int bn0 = blockIdx.x * 256;
    const int bm = bm0 + wm * 32;
    const int bn = bn0 + wn * 64;
    const unsigned arows = (unsigned)(M - bm0 < 64 ? M - bm0 : 64);
    const unsigned brows = (unsigned)(N - bn0 < 256 ? N - bn0 : 256);

    v8f acc[2][4];
    const v8f vzero = {0.f, 0.f, 0.f, 0.f, 0.f, 0.f, 0.f, 0.f};
#pragma unroll
    for (int i = 0; i < 2; ++i)
#pragma unroll
        for (int j = 0; j < 4; ++j) acc[i][j] = vzero;

    auto compute_tile = [&](const unsigned short (*Asb)[32],
                            const unsigned short (*Bsb)[32]) {
        Frag16 a[2], b[4];
#pragma unroll
        for (int i = 0; i < 2; ++i) {
            const unsigned short* p = &Asb[wm * 32 + i * 16 + lm][hi * 16];
            a[i].h[0] = *(const v8bf*)p;          // K = hi*16 + 0..7
            a[i].h[1] = *(const v8bf*)(p + 8);    // K = hi*16 + 8..15
        }
#pragma unroll
        for (int j = 0; j < 4; ++j)
            b[j].v = *(const v16bf*)&Bsb[wn * 64 + j * 16 + lm][hi * 16];
#pragma unroll
        for (int i = 0; i < 2; ++i)
#pragma unroll
            for (int j = 0; j < 4; ++j)
                acc[i][j] = __builtin_amdgcn_wmma_f32_16x16x32_bf16(
                    false, a[i].v, false, b[j].v, (short)0, acc[i][j], false, false);
    };

#if USE_TDM
    auto issue = [&](int buf, int kk) {
        tdm_load_tile_2d((unsigned)(uintptr_t)&As[buf][0][0],
                         A + (size_t)bm0 * lda + kk, 32, 64, 32, arows,
                         (unsigned long long)lda);
        tdm_load_tile_2d((unsigned)(uintptr_t)&Bs[buf][0][0],
                         B + (size_t)bn0 * ldb + kk, 32, 256, 32, brows,
                         (unsigned long long)ldb);
    };
    if (wave == 0) issue(0, 0);
    int buf = 0;
    for (int kk = 0; kk < K; kk += 32, buf ^= 1) {
        if (wave == 0) {
            if (kk + 32 < K) {
                issue(buf ^ 1, kk + 32);                       // prefetch next tiles
                __builtin_amdgcn_s_wait_tensorcnt(2);          // current tiles done
            } else {
                __builtin_amdgcn_s_wait_tensorcnt(0);
            }
        }
        __syncthreads();
        compute_tile(As[buf], Bs[buf]);
        __syncthreads();                                       // before overwriting buf^1
    }
#else
    Frag16 zf;
#pragma unroll
    for (int e = 0; e < 16; ++e) zf.u[e] = 0;
    for (int kk = 0; kk < K; kk += 32) {
        __syncthreads();
        {   // A tile: 64 rows x 32, 8 elems/thread
            const int r = threadIdx.x >> 2, c = (threadIdx.x & 3) * 8;
            const int gr = bm0 + r;
            *(v8bf*)&As[0][r][c] =
                (gr < M) ? *(const v8bf*)(A + (size_t)gr * lda + kk + c) : zf.h[0];
        }
        {   // B tile: 256 rows x 32, one full row (32 elems) per thread
            const int r = threadIdx.x;
            const int gr = bn0 + r;
            if (gr < N) {
                *(v16bf*)&Bs[0][r][0]  = *(const v16bf*)(B + (size_t)gr * ldb + kk);
                *(v16bf*)&Bs[0][r][16] = *(const v16bf*)(B + (size_t)gr * ldb + kk + 16);
            } else {
                *(v16bf*)&Bs[0][r][0]  = zf.v;
                *(v16bf*)&Bs[0][r][16] = zf.v;
            }
        }
        __syncthreads();
        compute_tile(As[0], Bs[0]);
    }
#endif

#pragma unroll
    for (int i = 0; i < 2; ++i) {
#pragma unroll
        for (int j = 0; j < 4; ++j) {
            const int col = bn + j * 16 + lm;
#pragma unroll
            for (int c8 = 0; c8 < 8; ++c8) {
                const int row = bm + i * 16 + hl + c8;   // C-frag: M = comp + hi*8
                if (row < M && col < N) {
                    float v = alpha * acc[i][j][c8];
                    if (bias) v += bias[col];
                    size_t idx = (size_t)row * ldc + col;
                    if (accumulate) v += out_ld(C[idx]);
                    out_st(C[idx], v);
                }
            }
        }
    }
}

// ---------- fused flash attention ----------
// qkv: (B*N, 3C) bf16. Block = one (b,h) x 128 queries (8 waves x 16 queries).
// Computes S^T = K_tile x Q^T so P lands directly in the WMMA A-matrix layout.
__global__ __launch_bounds__(256)
void attn_flash_wmma(const unsigned short* __restrict__ qkv,
                     unsigned short* __restrict__ out,
                     int Bb, int N, int C) {
    const int H = 16, D = 64;
    const int b = blockIdx.x / H, h = blockIdx.x % H;
    const int lane = threadIdx.x & 31;
    const int wave = threadIdx.x >> 5;
    const int lm = lane & 15;
    const int hi = lane >> 4;
    const int hl = hi * 8;
    const int qbase = blockIdx.y * 128 + wave * 16;
    const float scale = 0.125f;                       // D^-0.5

    const size_t row3 = (size_t)3 * C;
    const unsigned short* base = qkv + (size_t)b * N * row3;
    const unsigned short* Qp = base + (size_t)qbase * row3 + h * D;
    const unsigned short* Kp = base + C + h * D;
    const unsigned short* Vp = base + 2 * C + h * D;

    // Q as B-matrix fragments (lane = query, 16 contiguous d per half-wave)
    v16bf qf[2];
#pragma unroll
    for (int ks = 0; ks < 2; ++ks)
        qf[ks] = *(const v16bf*)(Qp + (size_t)lm * row3 + ks * 32 + hi * 16);

    const v8f vzero = {0.f, 0.f, 0.f, 0.f, 0.f, 0.f, 0.f, 0.f};
    v8f o[4];
#pragma unroll
    for (int fc = 0; fc < 4; ++fc) o[fc] = vzero;
    float m_run = -INFINITY, l_run = 0.f;

    __shared__ __align__(32) unsigned short VT[64][32];   // V^T tile: [feat][token]
    const int tt = threadIdx.x >> 3;                      // token 0..31 (staging role)
    const int f0 = (threadIdx.x & 7) * 8;                 // feat base (staging role)

    for (int t0 = 0; t0 < N; t0 += 32) {
        __syncthreads();
        {   // cooperative transpose of 32x64 V tile into LDS
            Frag16 tv; tv.h[0] = *(const v8bf*)(Vp + (size_t)(t0 + tt) * row3 + f0);
#pragma unroll
            for (int e = 0; e < 8; ++e) VT[f0 + e][tt] = tv.u[e];
        }
        if (t0 + 32 < N) {   // prefetch next K/V tiles (global_prefetch_b8)
            __builtin_prefetch(Kp + (size_t)(t0 + 32 + lm) * row3, 0, 0);
            __builtin_prefetch(Vp + (size_t)(t0 + 32 + tt) * row3 + f0, 0, 0);
        }
        __syncthreads();

        // S^T = K_tile(32x64) x Q^T : two 16-token sub-tiles, K-dim = d (2 steps of 32)
        v8f s[2];
#pragma unroll
        for (int ts = 0; ts < 2; ++ts) {
            const unsigned short* kr = Kp + (size_t)(t0 + ts * 16 + lm) * row3;
            v8f c = vzero;
#pragma unroll
            for (int ks = 0; ks < 2; ++ks) {
                Frag16 ka;
                const unsigned short* p = kr + ks * 32 + hi * 16;
                ka.h[0] = *(const v8bf*)p;
                ka.h[1] = *(const v8bf*)(p + 8);
                c = __builtin_amdgcn_wmma_f32_16x16x32_bf16(
                        false, ka.v, false, qf[ks], (short)0, c, false, false);
            }
            s[ts] = c;
        }

        // online softmax: lane = query; 16 scores per lane + partner lane (xor 16)
        float lmax = -INFINITY;
#pragma unroll
        for (int e = 0; e < 8; ++e) {
            s[0][e] *= scale; s[1][e] *= scale;
            lmax = fmaxf(lmax, fmaxf(s[0][e], s[1][e]));
        }
        lmax = fmaxf(lmax, __shfl_xor(lmax, 16, 32));
        const float m_new = fmaxf(m_run, lmax);
        const float r = __expf(m_run - m_new);

        Frag16 pa;                         // P in A-matrix layout (M=query, K=token)
        float lsum = 0.f;
#pragma unroll
        for (int e = 0; e < 8; ++e) {
            float p0 = __expf(s[0][e] - m_new);    // tokens (hi*8)+e
            float p1 = __expf(s[1][e] - m_new);    // tokens 16+(hi*8)+e
            lsum += p0 + p1;
            pa.u[e] = f2b_rne(p0);                 // A-frag K = hl + e
            pa.u[e + 8] = f2b_rne(p1);             // A-frag K = hl + 16 + e
        }
        lsum += __shfl_xor(lsum, 16, 32);
        l_run = l_run * r + lsum;
        m_run = m_new;

        // rescale O rows (C-frag comp c -> query hl+c); broadcast per-row factor
#pragma unroll
        for (int e = 0; e < 8; ++e) {
            float re = __shfl(r, hl + e, 32);
            o[0][e] *= re; o[1][e] *= re; o[2][e] *= re; o[3][e] *= re;
        }
        // O += P x V  (B-frag: lane = feat, tokens hl+0..7 and hl+16..23 per half-wave)
#pragma unroll
        for (int fc = 0; fc < 4; ++fc) {
            Frag16 vb;
            vb.h[0] = *(const v8bf*)&VT[fc * 16 + lm][hl];
            vb.h[1] = *(const v8bf*)&VT[fc * 16 + lm][hl + 16];
            o[fc] = __builtin_amdgcn_wmma_f32_16x16x32_bf16(
                        false, pa.v, false, vb.v, (short)0, o[fc], false, false);
        }
    }

    // epilogue: divide by l, scatter to (B,N,C) bf16
    const float inv = 1.0f / l_run;
    unsigned short* orow = out + ((size_t)b * N + qbase) * (size_t)C + h * D;
#pragma unroll
    for (int e = 0; e < 8; ++e) {
        const float ie = __shfl(inv, hl + e, 32);
        const int q = hl + e;
#pragma unroll
        for (int fc = 0; fc < 4; ++fc)
            orow[(size_t)q * C + fc * 16 + lm] = f2b_rne(o[fc][e] * ie);
    }
}

// ---------- host launcher ----------
extern "C" void kernel_launch(void* const* d_in, const int* in_sizes, int n_in,
                              void* d_out, int out_size, void* d_ws, size_t ws_size,
                              hipStream_t stream) {
    (void)in_sizes; (void)n_in; (void)out_size; (void)ws_size;
    const int Bb = 16, Nn = 1024, Cc = 1024, R = 64;
    const size_t MT = (size_t)Bb * Nn;                 // 16384 token rows

    const float* x     = (const float*)d_in[0];
    const float* Wqkv  = (const float*)d_in[1];
    const float* bqkv  = (const float*)d_in[2];
    const float* kA    = (const float*)d_in[3];
    const float* kB    = (const float*)d_in[4];
    const float* vA    = (const float*)d_in[5];
    const float* vB    = (const float*)d_in[6];
    const float* Wproj = (const float*)d_in[7];
    const float* bproj = (const float*)d_in[8];
    float* outp = (float*)d_out;

    char* w = (char*)d_ws;
    auto carve = [&](size_t bytes) -> unsigned short* {
        unsigned short* p = (unsigned short*)w;
        w += (bytes + 255) & ~(size_t)255;
        return p;
    };
    unsigned short* xb     = carve(MT * Cc * 2);
    unsigned short* wqkvb  = carve((size_t)3 * Cc * Cc * 2);
    unsigned short* kAb    = carve((size_t)R * Cc * 2);
    unsigned short* kBb    = carve((size_t)Cc * R * 2);
    unsigned short* vAb    = carve((size_t)R * Cc * 2);
    unsigned short* vBb    = carve((size_t)Cc * R * 2);
    unsigned short* wprojb = carve((size_t)Cc * Cc * 2);
    unsigned short* qkvb   = carve(MT * 3 * Cc * 2);
    unsigned short* tkb    = carve(MT * R * 2);
    unsigned short* tvb    = carve(MT * R * 2);
    unsigned short* attnb  = carve(MT * Cc * 2);

    auto cvt = [&](const float* s, unsigned short* d, long n) {
        int blocks = (int)((n + 1023) / 1024); if (blocks > 8192) blocks = 8192;
        cvt_f32_bf16<<<dim3(blocks), dim3(256), 0, stream>>>(s, d, n);
    };
    cvt(x, xb, (long)(MT * Cc));
    cvt(Wqkv, wqkvb, (long)3 * Cc * Cc);
    cvt(kA, kAb, (long)R * Cc);
    cvt(kB, kBb, (long)Cc * R);
    cvt(vA, vAb, (long)R * Cc);
    cvt(vB, vBb, (long)Cc * R);
    cvt(Wproj, wprojb, (long)Cc * Cc);

    const dim3 blk(256);
    const int mBlocks = (int)(MT / 64);                 // 256

    // qkv = x @ Wqkv^T + b      (N = 3072 -> 12 column blocks of 256)
    gemm_bf16_wmma<unsigned short><<<dim3(12, mBlocks), blk, 0, stream>>>(
        xb, Cc, wqkvb, Cc, qkvb, (long)3 * Cc, bqkv, (int)MT, 3 * Cc, Cc, 1.0f, 0);
    // LoRA down-projections: t = x @ A^T  (N = 64 -> 1 column block)
    gemm_bf16_wmma<unsigned short><<<dim3(1, mBlocks), blk, 0, stream>>>(
        xb, Cc, kAb, Cc, tkb, (long)R, nullptr, (int)MT, R, Cc, 1.0f, 0);
    gemm_bf16_wmma<unsigned short><<<dim3(1, mBlocks), blk, 0, stream>>>(
        xb, Cc, vAb, Cc, tvb, (long)R, nullptr, (int)MT, R, Cc, 1.0f, 0);
    // LoRA up-projections accumulate into k / v sections of qkv (N = 1024 -> 4 blocks)
    const float ls = 1.0f / 64.0f;                      // alpha / rank
    gemm_bf16_wmma<unsigned short><<<dim3(4, mBlocks), blk, 0, stream>>>(
        tkb, R, kBb, R, qkvb + Cc, (long)3 * Cc, nullptr, (int)MT, Cc, R, ls, 1);
    gemm_bf16_wmma<unsigned short><<<dim3(4, mBlocks), blk, 0, stream>>>(
        tvb, R, vBb, R, qkvb + 2 * Cc, (long)3 * Cc, nullptr, (int)MT, Cc, R, ls, 1);
    // fused flash attention
    attn_flash_wmma<<<dim3(Bb * 16, Nn / 128), blk, 0, stream>>>(qkvb, attnb, Bb, Nn, Cc);
    // out = attn @ Wproj^T + b  (fp32 output, N = 1024 -> 4 blocks)
    gemm_bf16_wmma<float><<<dim3(4, mBlocks), blk, 0, stream>>>(
        attnb, Cc, wprojb, Cc, outp, (long)Cc, bproj, (int)MT, Cc, Cc, 1.0f, 0);
}
